// CrystalHypergraphConv_17549236372285
// MI455X (gfx1250) — matrix operand order repaired and checked by reading, not compiled, not run on hardware
//
#include <hip/hip_runtime.h>
#include <hip/hip_bf16.h>

#define NN 50000
#define NH 100000
#define NE 800000
#define NG 512
#define IN_DIM 92
#define HD 64
#define HOUTD 128
#define HEDGE_DIM 35
#define ZD 163
#define KPZ 192
#define KPX 96

typedef __attribute__((ext_vector_type(16))) __bf16 v16bf;
typedef __attribute__((ext_vector_type(8)))  float  v8f;

// ---------------- workspace layout (bytes) ----------------
#define OFF_H     0UL          // h [NN,64] f32                       12.8 MB
#define OFF_HXS   12800000UL   // hedge sums/means [NH,64] f32 (25.6MB) ; aliased by zbf [NN,192] bf16 (19.2MB)
#define OFF_ZSUM  38400000UL   // z sums [NN,100] f32 (20MB) ; aliased by xbf [NN,96] bf16 (9.6MB)
#define OFF_HCNT  58400000UL   // [NH] f32
#define OFF_NCNT  58800000UL   // [NN] f32
#define OFF_WSH   59000000UL   // shuffled bf16 weights (embed 6144 + 6*12288 elems)
#define OFF_GSUM  59200000UL   // [NG,64] f32
#define OFF_GCNT  59331072UL   // [NG] f32
#define OFF_GACT  59333120UL   // [NG,128] f32

__device__ __forceinline__ __bf16 f2bf(float f) {
  unsigned u = __float_as_uint(f);
  unsigned r = u + 0x7FFFu + ((u >> 16) & 1u);   // round-to-nearest-even
  unsigned short hi = (unsigned short)(r >> 16);
  __bf16 o;
  __builtin_memcpy(&o, &hi, 2);
  return o;
}
__device__ __forceinline__ float softplus_f(float x) {
  return fmaxf(x, 0.f) + log1pf(__expf(-fabsf(x)));
}
__device__ __forceinline__ float sigmoid_f(float x) {
  return 1.f / (1.f + __expf(-x));
}
__device__ __forceinline__ void atomAddF(float* p, float v) {
  unsafeAtomicAdd(p, v);   // -> global_atomic_add_f32
}

// ---------------- prep kernels ----------------
__global__ void k_cvt_x(const float* __restrict__ x, __bf16* __restrict__ xbf) {
  int gid = blockIdx.x * blockDim.x + threadIdx.x;
  if (gid >= NN * KPX) return;
  int n = gid / KPX, c = gid - n * KPX;
  float v = (c < IN_DIM) ? x[n * IN_DIM + c] : 0.f;
  xbf[gid] = f2bf(v);
}

// Shuffle W[K,64] f32 into per-lane B-fragment layout:
// dst[t][c][lane][i] = W[t*32 + 16*(i>>3) + 8*(lane>>4) + (i&7)][c*16 + (lane&15)]
__global__ void k_shuffle_w(const float* __restrict__ W, int K,
                            __bf16* __restrict__ dst, int nkt) {
  int gid = blockIdx.x * blockDim.x + threadIdx.x;
  if (gid >= nkt * 2048) return;
  int i = gid & 15;
  int lane = (gid >> 4) & 31;
  int c = (gid >> 9) & 3;
  int t = gid >> 11;
  int k = t * 32 + ((i >> 3) << 4) + ((lane >> 4) << 3) + (i & 7);
  int col = c * 16 + (lane & 15);
  float v = (k < K) ? W[k * 64 + col] : 0.f;
  dst[gid] = f2bf(v);
}

// ---------------- WMMA fragment loaders ----------------
__device__ __forceinline__ v16bf load_a_frag(const __bf16* __restrict__ arow_kt, int half) {
  // 16-bit A 16x32 layout: lanes<16: K 0..7 & 16..23 ; lanes>=16: K 8..15 & 24..31
  v16bf a;
  const __bf16* p0 = arow_kt + half * 8;
  const __bf16* p1 = p0 + 16;
#pragma unroll
  for (int i = 0; i < 8; ++i) { a[i] = p0[i]; a[8 + i] = p1[i]; }
  return a;
}
__device__ __forceinline__ v16bf load_b_frag(const __bf16* __restrict__ wsh,
                                             int t, int c, int lane) {
  const __bf16* p = wsh + ((((t << 2) + c) << 5) + lane) * 16;
  v16bf b;
#pragma unroll
  for (int i = 0; i < 16; ++i) b[i] = p[i];
  return b;
}

// ---------------- embed: h = x @ W_embed + b ----------------
__global__ void k_embed(const __bf16* __restrict__ xbf, const __bf16* __restrict__ wsh,
                        const float* __restrict__ bias, float* __restrict__ h) {
  int wave = threadIdx.x >> 5, lane = threadIdx.x & 31;
  int tile = blockIdx.x * (blockDim.x >> 5) + wave;
  if (tile >= NN / 16) return;
  int m0 = tile * 16;
  int half = lane >> 4, lm = lane & 15;
  const __bf16* arow = xbf + (size_t)(m0 + lm) * KPX;
  v8f acc[4] = {};
#pragma unroll
  for (int t = 0; t < 3; ++t) {
    v16bf a = load_a_frag(arow + t * 32, half);
#pragma unroll
    for (int c = 0; c < 4; ++c) {
      v16bf b = load_b_frag(wsh, t, c, lane);
      acc[c] = __builtin_amdgcn_wmma_f32_16x16x32_bf16(
          false, a, false, b, (short)0, acc[c], false, false);
    }
  }
#pragma unroll
  for (int c = 0; c < 4; ++c) {
    int col = c * 16 + lm;
#pragma unroll
    for (int r = 0; r < 8; ++r) {
      int row = m0 + half * 8 + r;
      h[row * HD + col] = acc[c][r] + bias[col];
    }
  }
}

// ---------------- fused layer GEMMs + gate + residual (h updated in place) ----------------
__global__ void k_layer_gemm(const __bf16* __restrict__ zbf,
                             const __bf16* __restrict__ wfsh, const __bf16* __restrict__ wcsh,
                             const float* __restrict__ biasf, const float* __restrict__ biasc,
                             float* __restrict__ h) {
  int wave = threadIdx.x >> 5, lane = threadIdx.x & 31;
  int tile = blockIdx.x * (blockDim.x >> 5) + wave;
  if (tile >= NN / 16) return;
  int m0 = tile * 16;
  int half = lane >> 4, lm = lane & 15;
  const __bf16* arow = zbf + (size_t)(m0 + lm) * KPZ;
  v8f af[4] = {}, ac[4] = {};
#pragma unroll
  for (int t = 0; t < 6; ++t) {
    v16bf a = load_a_frag(arow + t * 32, half);
#pragma unroll
    for (int c = 0; c < 4; ++c) {
      v16bf bf_frag = load_b_frag(wfsh, t, c, lane);
      af[c] = __builtin_amdgcn_wmma_f32_16x16x32_bf16(
          false, a, false, bf_frag, (short)0, af[c], false, false);
      v16bf bc_frag = load_b_frag(wcsh, t, c, lane);
      ac[c] = __builtin_amdgcn_wmma_f32_16x16x32_bf16(
          false, a, false, bc_frag, (short)0, ac[c], false, false);
    }
  }
#pragma unroll
  for (int c = 0; c < 4; ++c) {
    int col = c * 16 + lm;
#pragma unroll
    for (int r = 0; r < 8; ++r) {
      int row = m0 + half * 8 + r;
      float zf = af[c][r] + biasf[col];
      float zc = ac[c][r] + biasc[col];
      float o = sigmoid_f(zf) * softplus_f(zc);
      float hv = h[row * HD + col];
      h[row * HD + col] = softplus_f(o + hv);
    }
  }
}

// ---------------- scatter kernels ----------------
__global__ void k_hedge_scatter(const float* __restrict__ h,
                                const int* __restrict__ nidx, const int* __restrict__ hidx,
                                float* __restrict__ hxs, float* __restrict__ hcnt) {
  int gid = blockIdx.x * blockDim.x + threadIdx.x;
  if (gid >= NE * 16) return;
  int e = gid >> 4, q = gid & 15, d = q * 4;
  int n = nidx[e], hg = hidx[e];
  float4 v = *reinterpret_cast<const float4*>(h + (size_t)n * HD + d);
  float* o = hxs + (size_t)hg * HD + d;
  atomAddF(o + 0, v.x); atomAddF(o + 1, v.y);
  atomAddF(o + 2, v.z); atomAddF(o + 3, v.w);
  if (q == 0) atomAddF(&hcnt[hg], 1.f);
}

__global__ void k_hedge_finalize(float* __restrict__ hxs, const float* __restrict__ hcnt) {
  int gid = blockIdx.x * blockDim.x + threadIdx.x;
  if (gid >= NH * HD) return;
  hxs[gid] = hxs[gid] / fmaxf(hcnt[gid >> 6], 1.f);
}

__global__ void k_node_scatter(const int* __restrict__ nidx, const int* __restrict__ hidx,
                               const float* __restrict__ hedge_attr,
                               const float* __restrict__ hxs,
                               float* __restrict__ z_sum, float* __restrict__ ncnt) {
  int gid = blockIdx.x * blockDim.x + threadIdx.x;
  if (gid >= NE * 25) return;
  int e = gid / 25, q = gid - e * 25;
  int d = q * 4;
  int n = nidx[e], hg = hidx[e];
  float* zp = z_sum + (size_t)n * 100;
#pragma unroll
  for (int j = 0; j < 4; ++j) {
    int dd = d + j;
    if (dd >= 99) break;
    float v = (dd < HEDGE_DIM) ? hedge_attr[(size_t)hg * HEDGE_DIM + dd]
                               : hxs[(size_t)hg * HD + (dd - HEDGE_DIM)];
    atomAddF(zp + dd, v);
  }
  if (q == 0) atomAddF(&ncnt[n], 1.f);
}

__global__ void k_z_finalize(const float* __restrict__ h, const float* __restrict__ z_sum,
                             const float* __restrict__ ncnt, __bf16* __restrict__ zbf) {
  int gid = blockIdx.x * blockDim.x + threadIdx.x;
  if (gid >= NN * KPZ) return;
  int n = gid / KPZ, c = gid - n * KPZ;
  float cnt = ncnt[n];
  float v = 0.f;
  if (c < HD)       v = (cnt > 0.f) ? h[(size_t)n * HD + c] : 0.f;   // mean of x_i copies
  else if (c < ZD)  v = z_sum[(size_t)n * 100 + (c - HD)] / fmaxf(cnt, 1.f);
  zbf[gid] = f2bf(v);
}

// ---------------- pooling + head ----------------
__global__ void k_pool_scatter(const float* __restrict__ h, const int* __restrict__ batch,
                               float* __restrict__ g_sum, float* __restrict__ gcnt) {
  int gid = blockIdx.x * blockDim.x + threadIdx.x;
  if (gid >= NN * 16) return;
  int n = gid >> 4, q = gid & 15, d = q * 4;
  int b = batch[n];
  float4 v = *reinterpret_cast<const float4*>(h + (size_t)n * HD + d);
  float* o = g_sum + (size_t)b * HD + d;
  atomAddF(o + 0, v.x); atomAddF(o + 1, v.y);
  atomAddF(o + 2, v.z); atomAddF(o + 3, v.w);
  if (q == 0) atomAddF(&gcnt[b], 1.f);
}

__global__ void k_head1(const float* __restrict__ g_sum, const float* __restrict__ gcnt,
                        const float* __restrict__ Wp, const float* __restrict__ bp,
                        float* __restrict__ gact) {
  int gid = blockIdx.x * blockDim.x + threadIdx.x;
  if (gid >= NG * HOUTD) return;
  int b = gid >> 7, j = gid & 127;
  float invc = 1.f / fmaxf(gcnt[b], 1.f);
  float acc = bp[j];
#pragma unroll 8
  for (int k = 0; k < HD; ++k)
    acc += (g_sum[b * HD + k] * invc) * Wp[k * HOUTD + j];
  gact[gid] = softplus_f(acc);
}

__global__ void k_head2(const float* __restrict__ gact, const float* __restrict__ Wo,
                        const float* __restrict__ bo, float* __restrict__ out) {
  int gid = blockIdx.x * blockDim.x + threadIdx.x;
  if (gid >= NG) return;
  float acc = bo[0];
#pragma unroll 8
  for (int k = 0; k < HOUTD; ++k) acc += gact[gid * HOUTD + k] * Wo[k];
  out[gid] = acc;
}

// ---------------- host ----------------
extern "C" void kernel_launch(void* const* d_in, const int* in_sizes, int n_in,
                              void* d_out, int out_size, void* d_ws, size_t ws_size,
                              hipStream_t stream) {
  const float* x          = (const float*)d_in[0];
  const float* hedge_attr = (const float*)d_in[1];
  const int*   node_idx   = (const int*)d_in[2];
  const int*   hedge_idx  = (const int*)d_in[3];
  const int*   batch      = (const int*)d_in[4];
  const float* W_embed    = (const float*)d_in[5];
  const float* b_embed    = (const float*)d_in[6];
  const float* Wf         = (const float*)d_in[7];
  const float* bf         = (const float*)d_in[8];
  const float* Wc         = (const float*)d_in[9];
  const float* bc         = (const float*)d_in[10];
  const float* Wp         = (const float*)d_in[11];
  const float* bp         = (const float*)d_in[12];
  const float* Wo         = (const float*)d_in[13];
  const float* bo         = (const float*)d_in[14];

  char* ws = (char*)d_ws;
  float*  h     = (float*)(ws + OFF_H);
  float*  hxs   = (float*)(ws + OFF_HXS);
  __bf16* zbf   = (__bf16*)(ws + OFF_HXS);   // alias (disjoint lifetime)
  float*  z_sum = (float*)(ws + OFF_ZSUM);
  __bf16* xbf   = (__bf16*)(ws + OFF_ZSUM);  // alias (disjoint lifetime)
  float*  hcnt  = (float*)(ws + OFF_HCNT);
  float*  ncnt  = (float*)(ws + OFF_NCNT);
  __bf16* wsh   = (__bf16*)(ws + OFF_WSH);
  float*  g_sum = (float*)(ws + OFF_GSUM);
  float*  gcnt  = (float*)(ws + OFF_GCNT);
  float*  gact  = (float*)(ws + OFF_GACT);

  __bf16* wsh_embed = wsh;                              // 3*2048 elems
  auto wsh_mat = [&](int m) { return wsh + 3 * 2048 + (size_t)m * 6 * 2048; };

  const int T = 256;
  // prep: convert x to padded bf16; shuffle all weights into B-fragment layout
  k_cvt_x<<<(NN * KPX) / T, T, 0, stream>>>(x, xbf);
  k_shuffle_w<<<(3 * 2048) / T, T, 0, stream>>>(W_embed, IN_DIM, wsh_embed, 3);
  for (int l = 0; l < 3; ++l) {
    k_shuffle_w<<<(6 * 2048) / T, T, 0, stream>>>(Wf + (size_t)l * ZD * 64, ZD, wsh_mat(l * 2 + 0), 6);
    k_shuffle_w<<<(6 * 2048) / T, T, 0, stream>>>(Wc + (size_t)l * ZD * 64, ZD, wsh_mat(l * 2 + 1), 6);
  }

  const int gemm_blocks = (NN / 16 + 7) / 8;  // 8 waves/block, 1 tile/wave
  k_embed<<<gemm_blocks, T, 0, stream>>>(xbf, wsh_embed, b_embed, h);

  for (int l = 0; l < 3; ++l) {
    hipMemsetAsync(hxs, 0, (size_t)NH * HD * 4, stream);
    hipMemsetAsync(hcnt, 0, (size_t)NH * 4, stream);
    k_hedge_scatter<<<(NE * 16) / T, T, 0, stream>>>(h, node_idx, hedge_idx, hxs, hcnt);
    k_hedge_finalize<<<(NH * HD) / T, T, 0, stream>>>(hxs, hcnt);

    hipMemsetAsync(z_sum, 0, (size_t)NN * 100 * 4, stream);
    hipMemsetAsync(ncnt, 0, (size_t)NN * 4, stream);
    k_node_scatter<<<(NE * 25) / T, T, 0, stream>>>(node_idx, hedge_idx, hedge_attr, hxs, z_sum, ncnt);
    k_z_finalize<<<(NN * KPZ) / T, T, 0, stream>>>(h, z_sum, ncnt, zbf);

    k_layer_gemm<<<gemm_blocks, T, 0, stream>>>(zbf, wsh_mat(l * 2), wsh_mat(l * 2 + 1),
                                                bf + l * 64, bc + l * 64, h);
  }

  hipMemsetAsync(g_sum, 0, (size_t)NG * HD * 4, stream);
  hipMemsetAsync(gcnt, 0, (size_t)NG * 4, stream);
  k_pool_scatter<<<(NN * 16) / T, T, 0, stream>>>(h, batch, g_sum, gcnt);
  k_head1<<<(NG * HOUTD) / T, T, 0, stream>>>(g_sum, gcnt, Wp, bp, gact);
  k_head2<<<(NG + T - 1) / T, T, 0, stream>>>(gact, Wo, bo, (float*)d_out);
}
